// MoELayer_90263032692926
// MI455X (gfx1250) — matrix operand order repaired and checked by reading, not compiled, not run on hardware
//
#include <hip/hip_runtime.h>
#include <hip/hip_bf16.h>

// MoE expert-choice layer for gfx1250 (MI455X), wave32 + WMMA bf16 path.
// Pipeline: router(softmax gates) -> per-expert radix top-C select ->
// GEMM1 (gather + x@W1 + gelu) -> GEMM2 (h@W2, scale, atomic scatter-add).
// GEMMs use v_wmma_f32_16x16x32_bf16 with a 2-term bf16 split of both
// operands (hi*hi + hi*lo + lo*hi) for ~2^-17 relative accuracy.

#define E_ 8
#define D_ 1024
#define F_ 4096
#define T_ 8192
#define C_ 2048

typedef __attribute__((ext_vector_type(16))) __bf16 v16bf;
typedef __attribute__((ext_vector_type(8)))  float  v8f;

union FragU { uint4 q[2]; v16bf v; };

__device__ __forceinline__ unsigned short f2bf(float f) {
  unsigned u = __float_as_uint(f);
  u += 0x7FFFu + ((u >> 16) & 1u);           // round-to-nearest-even
  return (unsigned short)(u >> 16);
}
__device__ __forceinline__ float bf2f(unsigned short h) {
  return __uint_as_float(((unsigned)h) << 16);
}
__device__ __forceinline__ void split2(float f, unsigned short& hi, unsigned short& lo) {
  hi = f2bf(f);
  lo = f2bf(f - bf2f(hi));
}

__device__ __forceinline__ v8f wmma_bf16(v16bf a, v16bf b, v8f c) {
  return __builtin_amdgcn_wmma_f32_16x16x32_bf16(false, a, false, b, (short)0, c, false, false);
}

// A fragment: 16x32 bf16, LDS tile layout [row][k], row-major, 2B elems.
// lanes 0-15: elems 0..7 = K 0..7, elems 8..15 = K 16..23 (half=0)
// lanes16-31: elems 0..7 = K 8..15, elems 8..15 = K 24..31 (half=1)
__device__ __forceinline__ v16bf load_fragA(const unsigned short* base, int row, int h) {
  FragU u;
  const char* p = (const char*)base + row * 64 + h * 16;
  u.q[0] = *(const uint4*)p;
  u.q[1] = *(const uint4*)(p + 32);
  return u.v;
}
// B fragment: 32x16 bf16, LDS tile layout [n][k] (transposed), same K order as A:
// lanes 0-15: K 0..7, 16..23; lanes 16-31: K 8..15, 24..31.
__device__ __forceinline__ v16bf load_fragB(const unsigned short* base, int col, int h) {
  FragU u;
  const char* p = (const char*)base + col * 64 + h * 16;
  u.q[0] = *(const uint4*)p;
  u.q[1] = *(const uint4*)(p + 32);
  return u.v;
}

// ---------------- router: gates = softmax(x @ Wg), stored [E][T] ------------
__global__ __launch_bounds__(256) void moe_router(
    const float* __restrict__ x, const float* __restrict__ Wg,
    float* __restrict__ scores) {
  const int lane = threadIdx.x & 31;
  const int t = blockIdx.x * 8 + (threadIdx.x >> 5);
  float acc[E_];
#pragma unroll
  for (int e = 0; e < E_; ++e) acc[e] = 0.0f;
  const float* xr = x + (size_t)t * D_;
  for (int i = 0; i < D_ / 32; ++i) {
    int d = i * 32 + lane;
    float xv = xr[d];
    const float* wr = Wg + d * E_;
#pragma unroll
    for (int e = 0; e < E_; ++e) acc[e] += xv * wr[e];
  }
#pragma unroll
  for (int off = 16; off >= 1; off >>= 1) {
#pragma unroll
    for (int e = 0; e < E_; ++e) acc[e] += __shfl_xor(acc[e], off, 32);
  }
  if (lane == 0) {
    float mx = acc[0];
#pragma unroll
    for (int e = 1; e < E_; ++e) mx = fmaxf(mx, acc[e]);
    float s = 0.0f;
#pragma unroll
    for (int e = 0; e < E_; ++e) { acc[e] = expf(acc[e] - mx); s += acc[e]; }
    float inv = 1.0f / s;
#pragma unroll
    for (int e = 0; e < E_; ++e) scores[(size_t)e * T_ + t] = acc[e] * inv;
  }
}

// -------- per-expert top-C: MSB-first radix select on positive floats ------
__global__ __launch_bounds__(1024) void moe_select(
    const float* __restrict__ scores, int* __restrict__ idxb,
    float* __restrict__ tvb) {
  const int e = blockIdx.x;
  const int tid = threadIdx.x;
  unsigned key[8];
#pragma unroll
  for (int j = 0; j < 8; ++j)
    key[j] = __float_as_uint(scores[(size_t)e * T_ + tid + j * 1024]);

  __shared__ unsigned scnt;
  if (tid == 0) scnt = 0;
  __syncthreads();

  unsigned prefix = 0, decided = 0;
  int k = C_;                               // rank of threshold value
  for (int b = 30; b >= 0; --b) {
    unsigned bit = 1u << b;
    unsigned local = 0;
#pragma unroll
    for (int j = 0; j < 8; ++j)
      if ((key[j] & decided) == prefix && (key[j] & bit)) local++;
    if (local) atomicAdd(&scnt, local);
    __syncthreads();
    unsigned cnt = scnt;
    __syncthreads();
    if (tid == 0) scnt = 0;
    __syncthreads();
    if (cnt >= (unsigned)k) prefix |= bit; else k -= (int)cnt;
    decided |= bit;
  }
  const unsigned vstar = prefix;            // C-th largest score (bit pattern)
  const int L = C_ - k;                     // # keys strictly greater

  __shared__ unsigned cgt, ceq;
  if (tid == 0) { cgt = 0; ceq = 0; }
  __syncthreads();
#pragma unroll
  for (int j = 0; j < 8; ++j) {
    unsigned u = key[j];
    int t = tid + j * 1024;
    int slot = -1;
    if (u > vstar) {
      slot = (int)atomicAdd(&cgt, 1u);
    } else if (u == vstar) {
      unsigned p = atomicAdd(&ceq, 1u);
      if ((int)p < k) slot = L + (int)p;
    }
    if (slot >= 0) {
      idxb[e * C_ + slot] = t;
      tvb[e * C_ + slot]  = __uint_as_float(u);
    }
  }
}

// ---------------- init output: zeros + constant aux_loss (=2.0) ------------
__global__ void init_out(float* __restrict__ out) {
  size_t i = (size_t)blockIdx.x * blockDim.x + threadIdx.x;
  if (i < (size_t)T_ * D_ / 4) {
    float4 z; z.x = 0.f; z.y = 0.f; z.z = 0.f; z.w = 0.f;
    ((float4*)out)[i] = z;
  }
  // aux_loss = E * sum_e mean_gate_e * (C/T) = 8 * 1 * 0.25 = 2.0 exactly.
  if (i == 0) out[(size_t)T_ * D_] = 2.0f;
}

// ---------------- GEMM1: h = gelu(x[idx] @ W1 + b1) ------------------------
__global__ __launch_bounds__(256) void moe_gemm1(
    const float* __restrict__ x, const float* __restrict__ W1,
    const float* __restrict__ b1, const int* __restrict__ idxb,
    float* __restrict__ hbuf) {
  const int e  = blockIdx.z;
  const int m0 = blockIdx.y * 128;
  const int n0 = blockIdx.x * 128;
  const int tid = threadIdx.x;

  __shared__ __align__(16) unsigned short Ah[128 * 32], Al[128 * 32];
  __shared__ __align__(16) unsigned short Bh[128 * 32], Bl[128 * 32];

  int arow[4];
#pragma unroll
  for (int it = 0; it < 4; ++it)
    arow[it] = idxb[e * C_ + m0 + ((it * 256 + tid) >> 3)];

  const int lane = tid & 31;
  const int w    = tid >> 5;
  const int wm   = (w >> 1) * 32;
  const int wn   = (w & 1) * 64;
  const int half = lane >> 4;
  const int lm   = lane & 15;

  v8f acc[2][4] = {};

  for (int kk = 0; kk < D_; kk += 32) {
#pragma unroll
    for (int it = 0; it < 4; ++it) {           // stage A: gathered x rows
      int lin = it * 256 + tid;
      int r = lin >> 3, c4 = lin & 7;
      float4 v = *(const float4*)(x + (size_t)arow[it] * D_ + kk + c4 * 4);
      unsigned short h0, h1, h2, h3, l0, l1, l2, l3;
      split2(v.x, h0, l0); split2(v.y, h1, l1);
      split2(v.z, h2, l2); split2(v.w, h3, l3);
      uint2 ph, pl;
      ph.x = (unsigned)h0 | ((unsigned)h1 << 16);
      ph.y = (unsigned)h2 | ((unsigned)h3 << 16);
      pl.x = (unsigned)l0 | ((unsigned)l1 << 16);
      pl.y = (unsigned)l2 | ((unsigned)l3 << 16);
      *(uint2*)(Ah + r * 32 + c4 * 4) = ph;
      *(uint2*)(Al + r * 32 + c4 * 4) = pl;
    }
#pragma unroll
    for (int it = 0; it < 4; ++it) {           // stage B: W1 transposed [n][k]
      int lin = it * 256 + tid;
      int k = lin >> 5, n4 = (lin & 31) * 4;
      float4 v = *(const float4*)(W1 + ((size_t)e * D_ + kk + k) * F_ + n0 + n4);
      unsigned short hh, ll;
      split2(v.x, hh, ll); Bh[(n4 + 0) * 32 + k] = hh; Bl[(n4 + 0) * 32 + k] = ll;
      split2(v.y, hh, ll); Bh[(n4 + 1) * 32 + k] = hh; Bl[(n4 + 1) * 32 + k] = ll;
      split2(v.z, hh, ll); Bh[(n4 + 2) * 32 + k] = hh; Bl[(n4 + 2) * 32 + k] = ll;
      split2(v.w, hh, ll); Bh[(n4 + 3) * 32 + k] = hh; Bl[(n4 + 3) * 32 + k] = ll;
    }
    __syncthreads();

    v16bf ah[2], al[2];
#pragma unroll
    for (int fm = 0; fm < 2; ++fm) {
      ah[fm] = load_fragA(Ah, wm + fm * 16 + lm, half);
      al[fm] = load_fragA(Al, wm + fm * 16 + lm, half);
    }
#pragma unroll
    for (int fn = 0; fn < 4; ++fn) {
      v16bf bh = load_fragB(Bh, wn + fn * 16 + lm, half);
      v16bf bl = load_fragB(Bl, wn + fn * 16 + lm, half);
#pragma unroll
      for (int fm = 0; fm < 2; ++fm) {
        acc[fm][fn] = wmma_bf16(ah[fm], bh, acc[fm][fn]);
        acc[fm][fn] = wmma_bf16(ah[fm], bl, acc[fm][fn]);
        acc[fm][fn] = wmma_bf16(al[fm], bh, acc[fm][fn]);
      }
    }
    __syncthreads();
  }

#pragma unroll
  for (int fm = 0; fm < 2; ++fm) {
#pragma unroll
    for (int fn = 0; fn < 4; ++fn) {
#pragma unroll
      for (int v = 0; v < 8; ++v) {
        int m = m0 + wm + fm * 16 + v + 8 * half;
        int n = n0 + wn + fn * 16 + lm;
        float val = acc[fm][fn][v] + b1[e * F_ + n];
        float c3 = val * val * val;                       // tanh-approx gelu
        float t  = tanhf(0.7978845608028654f * (val + 0.044715f * c3));
        hbuf[(size_t)(e * C_ + m) * F_ + n] = 0.5f * val * (1.0f + t);
      }
    }
  }
}

// -------- GEMM2: out[idx] += topval * (h @ W2 + b2), atomic scatter --------
__global__ __launch_bounds__(256) void moe_gemm2(
    const float* __restrict__ hbuf, const float* __restrict__ W2,
    const float* __restrict__ b2, const int* __restrict__ idxb,
    const float* __restrict__ tvb, float* __restrict__ out) {
  const int e  = blockIdx.z;
  const int m0 = blockIdx.y * 128;
  const int n0 = blockIdx.x * 128;
  const int tid = threadIdx.x;

  __shared__ __align__(16) unsigned short Ah[128 * 32], Al[128 * 32];
  __shared__ __align__(16) unsigned short Bh[128 * 32], Bl[128 * 32];

  const int lane = tid & 31;
  const int w    = tid >> 5;
  const int wm   = (w >> 1) * 32;
  const int wn   = (w & 1) * 64;
  const int half = lane >> 4;
  const int lm   = lane & 15;

  v8f acc[2][4] = {};

  for (int kk = 0; kk < F_; kk += 32) {
#pragma unroll
    for (int it = 0; it < 4; ++it) {           // stage A: h rows (no gather)
      int lin = it * 256 + tid;
      int r = lin >> 3, c4 = lin & 7;
      float4 v = *(const float4*)(hbuf + (size_t)(e * C_ + m0 + r) * F_ + kk + c4 * 4);
      unsigned short h0, h1, h2, h3, l0, l1, l2, l3;
      split2(v.x, h0, l0); split2(v.y, h1, l1);
      split2(v.z, h2, l2); split2(v.w, h3, l3);
      uint2 ph, pl;
      ph.x = (unsigned)h0 | ((unsigned)h1 << 16);
      ph.y = (unsigned)h2 | ((unsigned)h3 << 16);
      pl.x = (unsigned)l0 | ((unsigned)l1 << 16);
      pl.y = (unsigned)l2 | ((unsigned)l3 << 16);
      *(uint2*)(Ah + r * 32 + c4 * 4) = ph;
      *(uint2*)(Al + r * 32 + c4 * 4) = pl;
    }
#pragma unroll
    for (int it = 0; it < 4; ++it) {           // stage B: W2 transposed [n][k]
      int lin = it * 256 + tid;
      int k = lin >> 5, n4 = (lin & 31) * 4;
      float4 v = *(const float4*)(W2 + ((size_t)e * F_ + kk + k) * D_ + n0 + n4);
      unsigned short hh, ll;
      split2(v.x, hh, ll); Bh[(n4 + 0) * 32 + k] = hh; Bl[(n4 + 0) * 32 + k] = ll;
      split2(v.y, hh, ll); Bh[(n4 + 1) * 32 + k] = hh; Bl[(n4 + 1) * 32 + k] = ll;
      split2(v.z, hh, ll); Bh[(n4 + 2) * 32 + k] = hh; Bl[(n4 + 2) * 32 + k] = ll;
      split2(v.w, hh, ll); Bh[(n4 + 3) * 32 + k] = hh; Bl[(n4 + 3) * 32 + k] = ll;
    }
    __syncthreads();

    v16bf ah[2], al[2];
#pragma unroll
    for (int fm = 0; fm < 2; ++fm) {
      ah[fm] = load_fragA(Ah, wm + fm * 16 + lm, half);
      al[fm] = load_fragA(Al, wm + fm * 16 + lm, half);
    }
#pragma unroll
    for (int fn = 0; fn < 4; ++fn) {
      v16bf bh = load_fragB(Bh, wn + fn * 16 + lm, half);
      v16bf bl = load_fragB(Bl, wn + fn * 16 + lm, half);
#pragma unroll
      for (int fm = 0; fm < 2; ++fm) {
        acc[fm][fn] = wmma_bf16(ah[fm], bh, acc[fm][fn]);
        acc[fm][fn] = wmma_bf16(ah[fm], bl, acc[fm][fn]);
        acc[fm][fn] = wmma_bf16(al[fm], bh, acc[fm][fn]);
      }
    }
    __syncthreads();
  }

#pragma unroll
  for (int fm = 0; fm < 2; ++fm) {
#pragma unroll
    for (int fn = 0; fn < 4; ++fn) {
#pragma unroll
      for (int v = 0; v < 8; ++v) {
        int m = m0 + wm + fm * 16 + v + 8 * half;   // row within expert [0,C)
        int n = n0 + wn + fn * 16 + lm;
        float y = (acc[fm][fn][v] + b2[e * D_ + n]) * tvb[e * C_ + m];
        int tok = idxb[e * C_ + m];
        atomicAdd(out + (size_t)tok * D_ + n, y);
      }
    }
  }
}

extern "C" void kernel_launch(void* const* d_in, const int* in_sizes, int n_in,
                              void* d_out, int out_size, void* d_ws, size_t ws_size,
                              hipStream_t stream) {
  const float* x  = (const float*)d_in[0];
  const float* Wg = (const float*)d_in[1];
  const float* W1 = (const float*)d_in[2];
  const float* b1 = (const float*)d_in[3];
  const float* W2 = (const float*)d_in[4];
  const float* b2 = (const float*)d_in[5];
  float* out = (float*)d_out;

  char* ws = (char*)d_ws;
  float* scores = (float*)ws;  ws += (size_t)E_ * T_ * sizeof(float);
  int*   idxb   = (int*)ws;    ws += (size_t)E_ * C_ * sizeof(int);
  float* tvb    = (float*)ws;  ws += (size_t)E_ * C_ * sizeof(float);
  float* hbuf   = (float*)ws;  // E*C*F floats = 256 MB

  moe_router<<<dim3(T_ / 8), dim3(256), 0, stream>>>(x, Wg, scores);
  moe_select<<<dim3(E_), dim3(1024), 0, stream>>>(scores, idxb, tvb);
  init_out<<<dim3((unsigned)((size_t)T_ * D_ / 4 / 256)), dim3(256), 0, stream>>>(out);
  moe_gemm1<<<dim3(F_ / 128, C_ / 128, E_), dim3(256), 0, stream>>>(x, W1, b1, idxb, hbuf);
  moe_gemm2<<<dim3(D_ / 128, C_ / 128, E_), dim3(256), 0, stream>>>(hbuf, W2, b2, idxb, tvb, out);
}
